// Critic_13761075216952
// MI455X (gfx1250) — matrix operand order repaired and checked by reading, not compiled, not run on hardware
//
#include <hip/hip_runtime.h>

// ---------------------------------------------------------------------------
// CDNA5 / gfx1250 Critic: preprocessing MLP + 255-step LSTM + head.
//
// Split LSTM (cuDNN-style) when workspace allows:
//   Phase 1 (k_zpre):    Z_pre[t] = x_t @ Wk + bl   for all 255 steps
//                        -> fully parallel WMMA GEMM across 8160 WGs.
//   Phase 2 (k_lstm_rec): sequential, only h@Wrk per step (64 WMMA/wave/step),
//                        acc initialized from Z_pre (stored in D-reg layout).
// Fallback (k_lstm): fused x@Wk + h@Wrk persistent kernel (128 WMMA/wave/step).
// ---------------------------------------------------------------------------

typedef __attribute__((ext_vector_type(16))) __bf16 v16bf;
typedef __attribute__((ext_vector_type(8)))  __bf16 v8bf;
typedef __attribute__((ext_vector_type(8)))  float  v8f;

#define GMEM __attribute__((address_space(1)))

#if __has_builtin(__builtin_amdgcn_sched_barrier)
#define SCHED_FENCE() __builtin_amdgcn_sched_barrier(0)
#else
#define SCHED_FENCE()
#endif

#define B_TOT     512
#define STEPS     128
#define T_TOTAL   255          // 127 history + 128 action
#define D_IN      256
#define U         256          // LSTM units
#define FOURU     1024
#define BM        16           // batch rows per workgroup (one WMMA M-tile)
#define LSTRIDE   264          // LDS row stride in bf16 (256 + 8 pad)

// workspace byte offsets
#define OFF_WKT   ((size_t)0)                      // 1024*256 bf16 = 512 KB
#define OFF_WRKT  ((size_t)524288)                 // 512 KB
#define OFF_STATE ((size_t)1048576)                // 512*256 f32 = 512 KB
#define OFF_HFIN  ((size_t)1572864)                // 512*256 f32 = 512 KB
#define OFF_ACT   ((size_t)2097152)                // 512*768 f32 = 1.5 MB
#define OFF_Z     ((size_t)3670016)
// Z_pre: [255][32 btiles][8 waves][8 p][32 lanes][8 f32] = 534,773,760 B
#define ZTILE_BYTES ((size_t)65536)                // per (t,btile): 16x1024 f32
#define ZPRE_BYTES  ((size_t)T_TOTAL * 32 * ZTILE_BYTES)

__device__ __forceinline__ float sigm_f(float x) {
    return 1.0f / (1.0f + __expf(-x));
}
__device__ __forceinline__ float tanh_f(float x) {
    float ax = fabsf(x);
    float t  = __expf(-2.0f * ax);
    float r  = (1.0f - t) / (1.0f + t);
    return copysignf(r, x);
}
__device__ __forceinline__ v16bf cat8(v8bf lo, v8bf hi) {
    return __builtin_shufflevector(lo, hi, 0,1,2,3,4,5,6,7,8,9,10,11,12,13,14,15);
}

// ---------------------------------------------------------------------------
// Weight transpose + f32->bf16 convert: WT[n*256+k] = (bf16)W[k*1024+n]
// ---------------------------------------------------------------------------
__global__ void k_convert_weights(const float* __restrict__ Wk,
                                  const float* __restrict__ Wrk,
                                  void* __restrict__ wkT_raw,
                                  void* __restrict__ wrkT_raw) {
    __bf16* wkT  = (__bf16*)wkT_raw;
    __bf16* wrkT = (__bf16*)wrkT_raw;
    int idx = blockIdx.x * blockDim.x + threadIdx.x;   // 0 .. 262143
    int n = idx >> 8;
    int k = idx & 255;
    wkT[idx]  = (__bf16)Wk[k * FOURU + n];
    wrkT[idx] = (__bf16)Wrk[k * FOURU + n];
}

// ---------------------------------------------------------------------------
// Preprocess stage 1: act[b,0:768] = relu([ms | rs | re | im])
// ---------------------------------------------------------------------------
__global__ void k_prep1(const float* __restrict__ motion,
                        const float* __restrict__ robot,
                        const float* __restrict__ osr,
                        const float* __restrict__ osi,
                        const float* __restrict__ ore,
                        const float* __restrict__ oim,
                        const float* __restrict__ Wm, const float* __restrict__ bm,
                        const float* __restrict__ Wr, const float* __restrict__ br,
                        const float* __restrict__ Wre, const float* __restrict__ bre,
                        const float* __restrict__ Wim, const float* __restrict__ bim,
                        float* __restrict__ act) {
    int idx = blockIdx.x * blockDim.x + threadIdx.x;   // 0 .. 512*768-1
    int b = idx / 768;
    int n = idx - b * 768;
    float acc;
    if (n < 256) {
        acc = bm[n];
        for (int k = 0; k < 64; ++k) acc += motion[b * 64 + k] * Wm[k * 256 + n];
    } else if (n < 512) {
        int j = n - 256;
        acc = br[j];
        for (int k = 0; k < 128; ++k) acc += robot[b * 128 + k] * Wr[k * 256 + j];
    } else if (n < 640) {
        int j = n - 512;
        acc = bre[j];
        for (int k = 0; k < 64; ++k) acc += osr[b * 64 + k] * Wre[k * 128 + j];
        for (int k = 0; k < 64; ++k) acc += ore[b * 64 + k] * Wre[(64 + k) * 128 + j];
    } else {
        int j = n - 640;
        acc = bim[j];
        for (int k = 0; k < 64; ++k) acc += osi[b * 64 + k] * Wim[k * 128 + j];
        for (int k = 0; k < 64; ++k) acc += oim[b * 64 + k] * Wim[(64 + k) * 128 + j];
    }
    act[b * 768 + n] = fmaxf(acc, 0.0f);
}

// ---------------------------------------------------------------------------
// Preprocess stage 2: state[b,u] = relu(act[b,:] @ Wc[:,u] + bc[u])
// ---------------------------------------------------------------------------
__global__ void k_prep2(const float* __restrict__ act,
                        const float* __restrict__ Wc,
                        const float* __restrict__ bc,
                        float* __restrict__ state) {
    int idx = blockIdx.x * blockDim.x + threadIdx.x;   // 0 .. 512*256-1
    int b = idx >> 8;
    int u = idx & 255;
    float acc = bc[u];
    for (int k = 0; k < 768; ++k) acc += act[b * 768 + k] * Wc[k * 256 + u];
    state[idx] = fmaxf(acc, 0.0f);
}

// ---------------------------------------------------------------------------
// Phase 1: Z_pre[t,btile] = x_t @ Wk + bl, stored in D-register tile layout:
//   Z[((blk*8 + w)*8 + p)*32 + lane][0..7]   (blk = t*32 + btile)
// Grid: 255*32 WGs x 256 threads; fully parallel WMMA GEMM.
// ---------------------------------------------------------------------------
__global__ void __launch_bounds__(256, 1)
k_zpre(const float* __restrict__ history,
       const float* __restrict__ action,
       const void*  __restrict__ wkT_raw,
       const float* __restrict__ bl,
       float* __restrict__ Z) {
    __shared__ __bf16 xa[BM * LSTRIDE];

    const int tid  = threadIdx.x;
    const int w    = tid >> 5;
    const int lane = tid & 31;
    const int ln   = lane & 15;
    const int hi   = lane >> 4;
    const int blk  = blockIdx.x;       // t*32 + btile
    const int t    = blk >> 5;
    const int b0   = (blk & 31) * BM;

    const GMEM char* wkb = (const GMEM char*)(const char*)wkT_raw;

    // Stage x_t (bf16) into LDS.
    {
        const float* xs;
        int st;
        if (t < STEPS - 1) { xs = history; st = t; }
        else               { xs = action;  st = t - (STEPS - 1); }
        const float* src = xs + ((size_t)b0 * STEPS + st) * D_IN + tid;
        #pragma unroll
        for (int r = 0; r < BM; ++r)
            xa[r * LSTRIDE + tid] = (__bf16)src[(size_t)r * STEPS * D_IN];
    }
    __syncthreads();

    float bias[8];
    unsigned int boff[8];
    #pragma unroll
    for (int p = 0; p < 8; ++p) {
        unsigned int ncol = (unsigned)((p >> 1) * U + (w << 5) + ((p & 1) << 4) + ln);
        bias[p] = bl[ncol];
        boff[p] = ncol * (unsigned)(U * 2) + (unsigned)(hi << 5);
    }

    v8f acc[8];
    #pragma unroll
    for (int p = 0; p < 8; ++p)
        #pragma unroll
        for (int j = 0; j < 8; ++j) acc[p][j] = 0.0f;

    #pragma unroll 1
    for (int kt = 0; kt < 8; ++kt) {
        const int k0 = kt * 32;
        const __bf16* xp = &xa[ln * LSTRIDE + k0 + hi * 16];
        v16bf axv = cat8(*(const v8bf*)xp, *(const v8bf*)(xp + 8));
        // Batch all 8 B-tile loads; fence so the scheduler cannot sink them
        // back down to their consumers (pays L2 latency once per chunk).
        v16bf bt[8];
        #pragma unroll
        for (int p = 0; p < 8; ++p) {
            const GMEM v8bf* qk = (const GMEM v8bf*)(wkb + boff[p] + (unsigned)(kt << 6));
            bt[p] = cat8(qk[0], qk[1]);
        }
        SCHED_FENCE();
        #pragma unroll
        for (int p = 0; p < 8; ++p)
            acc[p] = __builtin_amdgcn_wmma_f32_16x16x32_bf16(
                false, axv, false, bt[p], (short)0, acc[p], false, false);
    }

    // Bias fold + non-temporal store (Z is a 534MB write-once stream; keep it
    // out of L2 so the hot 1MB weight copies stay resident).
    #pragma unroll
    for (int p = 0; p < 8; ++p) {
        #pragma unroll
        for (int j = 0; j < 8; ++j) acc[p][j] += bias[p];
        float* zo = Z + ((((size_t)blk * 8 + w) * 8 + p) * 32 + lane) * 8;
        __builtin_nontemporal_store(acc[p], (v8f*)zo);
    }
}

// ---------------------------------------------------------------------------
// Phase 2: sequential recurrence, z = Z_pre[t] + h@Wrk.
// 32 persistent WGs of 16 batch rows; 64 WMMA/wave/step.
// ---------------------------------------------------------------------------
__global__ void __launch_bounds__(256, 1)
k_lstm_rec(const float* __restrict__ Zpre,
           const float* __restrict__ state,
           const void*  __restrict__ wrkT_raw,
           float* __restrict__ hfinal) {
    __shared__ __bf16 ha[BM * LSTRIDE];

    const int tid  = threadIdx.x;
    const int w    = tid >> 5;
    const int lane = tid & 31;
    const int ln   = lane & 15;
    const int hi   = lane >> 4;
    const int b0   = blockIdx.x * BM;

    const GMEM char* wr_base = (const GMEM char*)(const char*)wrkT_raw;
    const GMEM char* z_base  = (const GMEM char*)(const char*)Zpre;

    #pragma unroll
    for (int r = 0; r < BM; ++r)
        ha[r * LSTRIDE + tid] = (__bf16)state[(b0 + r) * U + tid];

    unsigned int boff[8], zboff[8];
    #pragma unroll
    for (int p = 0; p < 8; ++p) {
        unsigned int ncol = (unsigned)((p >> 1) * U + (w << 5) + ((p & 1) << 4) + ln);
        boff[p]  = ncol * (unsigned)(U * 2) + (unsigned)(hi << 5);
        zboff[p] = (unsigned)((((w << 3) + p) << 5) + lane) << 5;   // *32 bytes
    }

    float cc[2][8], hreg[2][8];
    #pragma unroll
    for (int nt = 0; nt < 2; ++nt)
        #pragma unroll
        for (int j = 0; j < 8; ++j) {
            int row = b0 + j + (hi << 3);
            int col = (w << 5) + (nt << 4) + ln;
            cc[nt][j]   = state[row * U + col];
            hreg[nt][j] = 0.0f;
        }

    v8f acc[8];

    for (int t = 0; t < T_TOTAL; ++t) {
        __syncthreads();   // ha writes from previous step visible

        // acc <- Z_pre tile (x@Wk + bl, precomputed; D-register layout).
        const size_t zrow = ((size_t)t * 32 + (size_t)blockIdx.x) * ZTILE_BYTES;
        const GMEM char* zt = z_base + zrow;
        #pragma unroll
        for (int p = 0; p < 8; ++p)
            acc[p] = *(const GMEM v8f*)(zt + zboff[p]);

        // Prefetch next step's Z tiles so the post-barrier acc-init loads
        // (on the sequential critical path) hit cache.
        if (t + 1 < T_TOTAL) {
            const char* ztn = (const char*)Zpre + zrow + (size_t)32 * ZTILE_BYTES;
            #pragma unroll
            for (int p = 0; p < 8; ++p)
                __builtin_prefetch(ztn + zboff[p], 0, 0);
        }

        // Anti-LICM: opaque zero offset so Wrk tile loads stay in-loop.
        unsigned int zo = 0;
        asm volatile("" : "+s"(zo));
        const GMEM char* wrb = wr_base + zo;

        #pragma unroll 1
        for (int kt = 0; kt < 8; ++kt) {
            const int k0 = kt * 32;
            const __bf16* hp = &ha[ln * LSTRIDE + k0 + hi * 16];
            v16bf ahv = cat8(*(const v8bf*)hp, *(const v8bf*)(hp + 8));
            // Batch all 8 B-tile loads; fence against scheduler sinking.
            v16bf bt[8];
            #pragma unroll
            for (int p = 0; p < 8; ++p) {
                const GMEM v8bf* qr = (const GMEM v8bf*)(wrb + boff[p] + (unsigned)(kt << 6));
                bt[p] = cat8(qr[0], qr[1]);
            }
            SCHED_FENCE();
            #pragma unroll
            for (int p = 0; p < 8; ++p)
                acc[p] = __builtin_amdgcn_wmma_f32_16x16x32_bf16(
                    false, ahv, false, bt[p], (short)0, acc[p], false, false);
        }

        __syncthreads();   // all waves done reading ha

        #pragma unroll
        for (int nt = 0; nt < 2; ++nt) {
            #pragma unroll
            for (int j = 0; j < 8; ++j) {
                float zi = acc[0 * 2 + nt][j];
                float zf = acc[1 * 2 + nt][j];
                float zg = acc[2 * 2 + nt][j];
                float zoo = acc[3 * 2 + nt][j];
                float ig = sigm_f(zi);
                float fg = sigm_f(zf);
                float gg = tanh_f(zg);
                float og = sigm_f(zoo);
                float c  = fg * cc[nt][j] + ig * gg;
                cc[nt][j] = c;
                float h  = og * tanh_f(c);
                hreg[nt][j] = h;
                ha[(j + (hi << 3)) * LSTRIDE + (w << 5) + (nt << 4) + ln] = (__bf16)h;
            }
        }
    }

    #pragma unroll
    for (int nt = 0; nt < 2; ++nt)
        #pragma unroll
        for (int j = 0; j < 8; ++j) {
            int row = b0 + j + (hi << 3);
            int col = (w << 5) + (nt << 4) + ln;
            hfinal[row * U + col] = hreg[nt][j];
        }
}

// ---------------------------------------------------------------------------
// Fused fallback LSTM (when workspace cannot hold Z_pre): x@Wk + h@Wrk.
// ---------------------------------------------------------------------------
__global__ void __launch_bounds__(256, 1)
k_lstm(const float* __restrict__ history,
       const float* __restrict__ action,
       const float* __restrict__ state,
       const void*  __restrict__ wkT_raw,
       const void*  __restrict__ wrkT_raw,
       const float* __restrict__ bl,
       float* __restrict__ hfinal) {
    __shared__ __bf16 xa[BM * LSTRIDE];
    __shared__ __bf16 ha[BM * LSTRIDE];

    const int tid  = threadIdx.x;
    const int w    = tid >> 5;
    const int lane = tid & 31;
    const int ln   = lane & 15;
    const int hi   = lane >> 4;
    const int b0   = blockIdx.x * BM;

    const GMEM char* wk_base = (const GMEM char*)(const char*)wkT_raw;
    const GMEM char* wr_base = (const GMEM char*)(const char*)wrkT_raw;

    #pragma unroll
    for (int r = 0; r < BM; ++r)
        ha[r * LSTRIDE + tid] = (__bf16)state[(b0 + r) * U + tid];

    float bias[8];
    unsigned int boff[8];
    #pragma unroll
    for (int p = 0; p < 8; ++p) {
        unsigned int ncol = (unsigned)((p >> 1) * U + (w << 5) + ((p & 1) << 4) + ln);
        bias[p] = bl[ncol];
        boff[p] = ncol * (unsigned)(U * 2) + (unsigned)(hi << 5);
    }

    float cc[2][8], hreg[2][8];
    #pragma unroll
    for (int nt = 0; nt < 2; ++nt)
        #pragma unroll
        for (int j = 0; j < 8; ++j) {
            int row = b0 + j + (hi << 3);
            int col = (w << 5) + (nt << 4) + ln;
            cc[nt][j]   = state[row * U + col];
            hreg[nt][j] = 0.0f;
        }

    v8f acc[8];

    for (int t = 0; t < T_TOTAL; ++t) {
        __syncthreads();
        {
            const float* xs;
            int st;
            if (t < STEPS - 1) { xs = history; st = t; }
            else               { xs = action;  st = t - (STEPS - 1); }
            const float* src = xs + ((size_t)b0 * STEPS + st) * D_IN + tid;
            #pragma unroll
            for (int r = 0; r < BM; ++r)
                xa[r * LSTRIDE + tid] = (__bf16)src[(size_t)r * STEPS * D_IN];
        }
        __syncthreads();

        unsigned int zo = 0;
        asm volatile("" : "+s"(zo));
        const GMEM char* wkb = wk_base + zo;
        const GMEM char* wrb = wr_base + zo;

        #pragma unroll
        for (int p = 0; p < 8; ++p)
            #pragma unroll
            for (int j = 0; j < 8; ++j) acc[p][j] = 0.0f;

        #pragma unroll 1
        for (int kt = 0; kt < 8; ++kt) {
            const int k0 = kt * 32;
            const __bf16* xp = &xa[ln * LSTRIDE + k0 + hi * 16];
            v16bf axv = cat8(*(const v8bf*)xp, *(const v8bf*)(xp + 8));
            const __bf16* hp = &ha[ln * LSTRIDE + k0 + hi * 16];
            v16bf ahv = cat8(*(const v8bf*)hp, *(const v8bf*)(hp + 8));

            v16bf bt[8];
            #pragma unroll
            for (int p = 0; p < 8; ++p) {
                const GMEM v8bf* qk = (const GMEM v8bf*)(wkb + boff[p] + (unsigned)(kt << 6));
                bt[p] = cat8(qk[0], qk[1]);
            }
            SCHED_FENCE();
            #pragma unroll
            for (int p = 0; p < 8; ++p)
                acc[p] = __builtin_amdgcn_wmma_f32_16x16x32_bf16(
                    false, axv, false, bt[p], (short)0, acc[p], false, false);

            #pragma unroll
            for (int p = 0; p < 8; ++p) {
                const GMEM v8bf* qr = (const GMEM v8bf*)(wrb + boff[p] + (unsigned)(kt << 6));
                bt[p] = cat8(qr[0], qr[1]);
            }
            SCHED_FENCE();
            #pragma unroll
            for (int p = 0; p < 8; ++p)
                acc[p] = __builtin_amdgcn_wmma_f32_16x16x32_bf16(
                    false, ahv, false, bt[p], (short)0, acc[p], false, false);
        }

        __syncthreads();

        #pragma unroll
        for (int nt = 0; nt < 2; ++nt) {
            #pragma unroll
            for (int j = 0; j < 8; ++j) {
                float zi = acc[0 * 2 + nt][j] + bias[0 * 2 + nt];
                float zf = acc[1 * 2 + nt][j] + bias[1 * 2 + nt];
                float zg = acc[2 * 2 + nt][j] + bias[2 * 2 + nt];
                float zoo = acc[3 * 2 + nt][j] + bias[3 * 2 + nt];
                float ig = sigm_f(zi);
                float fg = sigm_f(zf);
                float gg = tanh_f(zg);
                float og = sigm_f(zoo);
                float c  = fg * cc[nt][j] + ig * gg;
                cc[nt][j] = c;
                float h  = og * tanh_f(c);
                hreg[nt][j] = h;
                ha[(j + (hi << 3)) * LSTRIDE + (w << 5) + (nt << 4) + ln] = (__bf16)h;
            }
        }
    }

    #pragma unroll
    for (int nt = 0; nt < 2; ++nt)
        #pragma unroll
        for (int j = 0; j < 8; ++j) {
            int row = b0 + j + (hi << 3);
            int col = (w << 5) + (nt << 4) + ln;
            hfinal[row * U + col] = hreg[nt][j];
        }
}

// ---------------------------------------------------------------------------
// Output head: out[b] = relu(h[b,:] @ Wo + bo)
// ---------------------------------------------------------------------------
__global__ void k_out(const float* __restrict__ hfinal,
                      const float* __restrict__ Wo,
                      const float* __restrict__ bo,
                      float* __restrict__ out) {
    int b = blockIdx.x * blockDim.x + threadIdx.x;   // 0..511
    float acc = bo[0];
    for (int k = 0; k < U; ++k) acc += hfinal[b * U + k] * Wo[k];
    out[b] = fmaxf(acc, 0.0f);
}

// ---------------------------------------------------------------------------
extern "C" void kernel_launch(void* const* d_in, const int* in_sizes, int n_in,
                              void* d_out, int out_size, void* d_ws, size_t ws_size,
                              hipStream_t stream) {
    (void)in_sizes; (void)n_in; (void)out_size;

    const float* motion  = (const float*)d_in[0];
    const float* robot   = (const float*)d_in[1];
    const float* osr     = (const float*)d_in[2];
    const float* osi     = (const float*)d_in[3];
    const float* history = (const float*)d_in[4];
    const float* action  = (const float*)d_in[5];
    const float* ore     = (const float*)d_in[6];
    const float* oim     = (const float*)d_in[7];
    const float* Wm  = (const float*)d_in[8];
    const float* bm  = (const float*)d_in[9];
    const float* Wr  = (const float*)d_in[10];
    const float* br  = (const float*)d_in[11];
    const float* Wre = (const float*)d_in[12];
    const float* bre = (const float*)d_in[13];
    const float* Wim = (const float*)d_in[14];
    const float* bim = (const float*)d_in[15];
    const float* Wc  = (const float*)d_in[16];
    const float* bc  = (const float*)d_in[17];
    const float* Wk  = (const float*)d_in[18];
    const float* Wrk = (const float*)d_in[19];
    const float* bl  = (const float*)d_in[20];
    const float* Wo  = (const float*)d_in[21];
    const float* bo  = (const float*)d_in[22];

    char* ws = (char*)d_ws;
    void*  wkT   = (void*)(ws + OFF_WKT);
    void*  wrkT  = (void*)(ws + OFF_WRKT);
    float* state = (float*)(ws + OFF_STATE);
    float* hfin  = (float*)(ws + OFF_HFIN);
    float* act   = (float*)(ws + OFF_ACT);
    float* Zpre  = (float*)(ws + OFF_Z);

    // 1) bf16 transposed weight copies
    k_convert_weights<<<1024, 256, 0, stream>>>(Wk, Wrk, wkT, wrkT);

    // 2) preprocessing MLP -> state
    k_prep1<<<(B_TOT * 768) / 256, 256, 0, stream>>>(
        motion, robot, osr, osi, ore, oim,
        Wm, bm, Wr, br, Wre, bre, Wim, bim, act);
    k_prep2<<<(B_TOT * 256) / 256, 256, 0, stream>>>(act, Wc, bc, state);

    // 3) LSTM
    if (ws_size >= OFF_Z + ZPRE_BYTES) {
        // Split path: full-chip parallel input GEMM + lean recurrence.
        k_zpre<<<T_TOTAL * 32, 256, 0, stream>>>(history, action, wkT, bl, Zpre);
        k_lstm_rec<<<B_TOT / BM, 256, 0, stream>>>(Zpre, state, wrkT, hfin);
    } else {
        // Fused fallback.
        k_lstm<<<B_TOT / BM, 256, 0, stream>>>(history, action, state,
                                               wkT, wrkT, bl, hfin);
    }

    // 4) output head
    k_out<<<B_TOT / 256, 256, 0, stream>>>(hfin, Wo, bo, (float*)d_out);
}
